// ProgramAdapter_64596308132166
// MI455X (gfx1250) — matrix operand order repaired and checked by reading, not compiled, not run on hardware
//
#include <hip/hip_runtime.h>
#include <hip/hip_bf16.h>

#define B_ 8
#define S_ 2048
#define H_ 512
#define P_ 16
#define K_ 256
#define T_ (B_*S_)

typedef __attribute__((ext_vector_type(16))) __bf16 v16bf;
typedef __attribute__((ext_vector_type(8)))  __bf16 v8bf;
typedef __attribute__((ext_vector_type(8)))  float  v8f;

__device__ __forceinline__ __bf16 f2bf(float f) {
  union { float f; unsigned u; } in; in.f = f;
  unsigned r = in.u + 0x7FFFu + ((in.u >> 16) & 1u);   // round-to-nearest-even
  union { unsigned short s; __bf16 b; } o; o.s = (unsigned short)(r >> 16);
  return o.b;
}

__device__ __forceinline__ v8f v8f_zero() {
  v8f z = {0.f,0.f,0.f,0.f,0.f,0.f,0.f,0.f};
  return z;
}

// CDNA5 async copy: global -> LDS (16B), tracked with ASYNCcnt.
// %0 = per-lane LDS byte offset (VGPR), %1 = 64-bit global address (VGPR pair).
__device__ __forceinline__ void async_g2l_b128(unsigned lds_off, const void* gaddr) {
  asm volatile("global_load_async_to_lds_b128 %0, %1, off"
               :: "v"(lds_off), "v"(gaddr) : "memory");
}
__device__ __forceinline__ void wait_asynccnt0() {
  asm volatile("s_wait_asynccnt 0" ::: "memory");
}

// A-fragment (16x32 bf16) from LDS, row-major tile with leading dim `ld`.
// Lane r=lane&15 holds row m0+r; lo-half lanes take K groups {0..7,16..23},
// hi-half lanes take {8..15,24..31} (ISA 16-bit A layout).
__device__ __forceinline__ v16bf afrag_lds(const __bf16* tile, int ld, int m0, int k0, int lane) {
  int r  = lane & 15;
  int hi = (lane >> 4) * 8;
  const __bf16* p = tile + (m0 + r) * ld + k0 + hi;
  union { v16bf v; v8bf h[2]; } u;
  u.h[0] = *(const v8bf*)(p);        // K = k0+hi .. k0+hi+7
  u.h[1] = *(const v8bf*)(p + 16);   // K = k0+16+hi .. +7
  return u.v;
}

// B-fragment (32x16 bf16) from global, weights stored transposed [N][Kdim]
// so each lane (column n) reads K groups {hi..hi+7, 16+hi..16+hi+7}, hi = 0 or 8.
__device__ __forceinline__ v16bf bfrag_gl(const __bf16* base, int ld, int n0, int k0, int lane) {
  int n  = n0 + (lane & 15);
  int hi = (lane >> 4) * 8;
  const __bf16* p = base + (size_t)n * ld + k0 + hi;
  union { v16bf v; v8bf h[2]; } u;
  u.h[0] = *(const v8bf*)(p);
  u.h[1] = *(const v8bf*)(p + 16);
  return u.v;
}

__device__ __forceinline__ v8f wmma_bf16(v16bf a, v16bf b, v8f c) {
  return __builtin_amdgcn_wmma_f32_16x16x32_bf16(false, a, false, b, (short)0, c, false, false);
}

// ---------------- routing: pooled mean -> linear head -> softmax ----------------
__global__ __launch_bounds__(256) void k_router(const float* __restrict__ hidden,
                                                const float* __restrict__ head_w,
                                                const float* __restrict__ head_b,
                                                float* __restrict__ probs) {
  __shared__ float pooled[H_];
  __shared__ float lg[P_];
  const int b = blockIdx.x, tid = threadIdx.x;
  for (int h = tid; h < H_; h += 256) {
    float s = 0.f;
    const float* src = hidden + (size_t)b * S_ * H_ + h;
    for (int ss = 0; ss < S_; ++ss) s += src[(size_t)ss * H_];
    pooled[h] = s * (1.0f / (float)S_);
  }
  __syncthreads();
  if (tid < P_) {
    float l = head_b[tid];
    for (int h = 0; h < H_; ++h) l += pooled[h] * head_w[tid * H_ + h];
    lg[tid] = l;
  }
  __syncthreads();
  if (tid == 0) {
    float m = lg[0];
    for (int p = 1; p < P_; ++p) m = fmaxf(m, lg[p]);
    float e[P_]; float sum = 0.f;
    for (int p = 0; p < P_; ++p) { e[p] = expf(lg[p] - m); sum += e[p]; }
    float inv = 1.f / sum;
    for (int p = 0; p < P_; ++p) probs[b * P_ + p] = e[p] * inv;
  }
}

// ---------------- b1_eff[p,k] = b1[p,k] + embeds[p,:] @ w1[p,:,k] ----------------
__global__ __launch_bounds__(256) void k_b1eff(const float* __restrict__ embeds,
                                               const float* __restrict__ w1,
                                               const float* __restrict__ b1,
                                               float* __restrict__ b1e) {
  const int p = blockIdx.x, k = threadIdx.x;
  float s = b1[p * K_ + k];
  const float* e = embeds + p * H_;
  const float* w = w1 + (size_t)p * H_ * K_ + k;
  for (int h = 0; h < H_; ++h) s += e[h] * w[(size_t)h * K_];
  b1e[p * K_ + k] = s;
}

// ---------------- transpose+convert weights: dst[p][c][r] = bf16(src[p][r][c]) ----------------
__global__ __launch_bounds__(256) void k_transpose_bf16(const float* __restrict__ src,
                                                        __bf16* __restrict__ dst,
                                                        int R, int C) {
  size_t idx = (size_t)blockIdx.x * 256 + threadIdx.x;  // linear over dst [P][C][R]
  int r = (int)(idx % R);
  size_t t = idx / R;
  int c = (int)(t % C);
  int p = (int)(t / C);
  dst[idx] = f2bf(src[((size_t)p * R + r) * C + c]);
}

// ---------------- hidden f32 -> bf16 ----------------
__global__ __launch_bounds__(256) void k_hid_bf16(const float* __restrict__ hidden,
                                                  __bf16* __restrict__ dst) {
  size_t i = (size_t)blockIdx.x * 256 + threadIdx.x;    // 4 elements per thread
  float4 v = ((const float4*)hidden)[i];
  union { __bf16 b[4]; ushort4 u; } o;
  o.b[0] = f2bf(v.x); o.b[1] = f2bf(v.y); o.b[2] = f2bf(v.z); o.b[3] = f2bf(v.w);
  ((ushort4*)dst)[i] = o.u;
}

// ---------------- main fused expert-MLP kernel ----------------
// One block = 32 tokens x full H. 8 waves. Per expert p:
//   GEMM1: mid(32x256) = probs[b,p] * gelu(hid(32x512) @ w1t + b1_eff)
//   GEMM2: acc(32x512) += mid @ w2t      (WMMA C-operand accumulation)
#define LDH 520   // 512 + 8 pad (16B), keeps A-frag LDS reads conflict-free
#define LDM 264   // 256 + 8 pad

__global__ __launch_bounds__(256) void k_mlp(const __bf16* __restrict__ hidb,
                                             const __bf16* __restrict__ w1t,
                                             const __bf16* __restrict__ w2t,
                                             const float* __restrict__ b1e,
                                             const float* __restrict__ b2,
                                             const float* __restrict__ probs,
                                             float* __restrict__ out) {
  __shared__ __bf16 sh_hid[32 * LDH];
  __shared__ __bf16 sh_mid[32 * LDM];
  __shared__ float  sh_bt[H_];

  const int tid  = threadIdx.x;
  const int lane = tid & 31;
  const int wv   = tid >> 5;            // wave 0..7
  const int t0   = blockIdx.x * 32;     // token tile base
  const int b    = t0 >> 11;            // t0 / S_

  // stage 32x512 bf16 hidden tile into LDS via async global->LDS copies
  {
    int row = tid >> 3;
    int seg = (tid & 7) * 64;           // 64 bf16 = 128B per thread
    const __bf16* src = hidb + (size_t)(t0 + row) * H_ + seg;
    unsigned dst_off = (unsigned)(unsigned long long)(sh_hid + row * LDH + seg);
    #pragma unroll
    for (int i = 0; i < 8; ++i)
      async_g2l_b128(dst_off + i * 16, (const void*)(src + i * 8));
  }
  // bias term bterm[h] = sum_p probs[b,p] * b2[p,h]
  for (int h = tid; h < H_; h += 256) {
    float s = 0.f;
    #pragma unroll
    for (int pp = 0; pp < P_; ++pp) s += probs[b * P_ + pp] * b2[pp * H_ + h];
    sh_bt[h] = s;
  }
  wait_asynccnt0();
  __syncthreads();

  v8f acc[8];
  #pragma unroll
  for (int t = 0; t < 8; ++t) acc[t] = v8f_zero();

  const int n1 = wv * 32;   // GEMM1: this wave owns output cols [n1, n1+32)
  const int n2 = wv * 64;   // GEMM2: this wave owns output cols [n2, n2+64)

  for (int p = 0; p < P_; ++p) {
    const __bf16* w1p = w1t + (size_t)p * K_ * H_;   // [K_][H_]
    const __bf16* w2p = w2t + (size_t)p * H_ * K_;   // [H_][K_]
    const float prob = probs[b * P_ + p];

    // ---------- GEMM1 ----------
    v8f c1[4];
    #pragma unroll
    for (int t = 0; t < 4; ++t) c1[t] = v8f_zero();
    for (int kk = 0; kk < H_; kk += 32) {
      if (kk + 32 < H_)
        __builtin_prefetch((const void*)(w1p + (size_t)(n1 + (lane & 15)) * H_ + kk + 32), 0, 1);
      v16bf a0 = afrag_lds(sh_hid, LDH, 0,  kk, lane);
      v16bf a1 = afrag_lds(sh_hid, LDH, 16, kk, lane);
      v16bf bl = bfrag_gl(w1p, H_, n1,      kk, lane);
      v16bf bh = bfrag_gl(w1p, H_, n1 + 16, kk, lane);
      c1[0] = wmma_bf16(a0, bl, c1[0]);
      c1[1] = wmma_bf16(a0, bh, c1[1]);
      c1[2] = wmma_bf16(a1, bl, c1[2]);
      c1[3] = wmma_bf16(a1, bh, c1[3]);
    }
    // epilogue: +bias, exact GELU, pre-scale by prob, store bf16 mid tile
    #pragma unroll
    for (int t = 0; t < 4; ++t) {
      int m0 = (t >> 1) * 16;
      int nc = n1 + (t & 1) * 16 + (lane & 15);
      int rb = m0 + ((lane >> 4) ? 8 : 0);
      float bias = b1e[p * K_ + nc];
      #pragma unroll
      for (int j = 0; j < 8; ++j) {
        float x = c1[t][j] + bias;
        float g = 0.5f * x * (1.f + erff(x * 0.70710678118f));
        sh_mid[(rb + j) * LDM + nc] = f2bf(prob * g);
      }
    }
    __syncthreads();

    // ---------- GEMM2: direct accumulation into persistent C ----------
    for (int kk = 0; kk < K_; kk += 32) {
      v16bf a0 = afrag_lds(sh_mid, LDM, 0,  kk, lane);
      v16bf a1 = afrag_lds(sh_mid, LDM, 16, kk, lane);
      #pragma unroll
      for (int nt = 0; nt < 4; ++nt) {
        v16bf bb = bfrag_gl(w2p, K_, n2 + nt * 16, kk, lane);
        acc[nt]     = wmma_bf16(a0, bb, acc[nt]);
        acc[4 + nt] = wmma_bf16(a1, bb, acc[4 + nt]);
      }
    }
    __syncthreads();   // protect sh_mid before next expert overwrites it
  }

  // ---------- writeout: out = acc + bterm ----------
  #pragma unroll
  for (int t = 0; t < 8; ++t) {
    int m0  = (t >> 2) * 16;
    int nt  = t & 3;
    int col = n2 + nt * 16 + (lane & 15);
    int rb  = m0 + ((lane >> 4) ? 8 : 0);
    float bt = sh_bt[col];
    #pragma unroll
    for (int j = 0; j < 8; ++j)
      out[(size_t)(t0 + rb + j) * H_ + col] = acc[t][j] + bt;
  }
}

extern "C" void kernel_launch(void* const* d_in, const int* in_sizes, int n_in,
                              void* d_out, int out_size, void* d_ws, size_t ws_size,
                              hipStream_t stream) {
  const float* hidden = (const float*)d_in[0];
  const float* head_w = (const float*)d_in[1];
  const float* head_b = (const float*)d_in[2];
  const float* embeds = (const float*)d_in[3];
  const float* w1     = (const float*)d_in[4];
  const float* b1     = (const float*)d_in[5];
  const float* w2     = (const float*)d_in[6];
  const float* b2     = (const float*)d_in[7];
  float* out = (float*)d_out;

  char* ws = (char*)d_ws;
  float*  probs_ws = (float*)ws;                                  // B_*P_ f32
  float*  b1e_ws   = (float*)(ws + 1024);                         // P_*K_ f32
  __bf16* w1t      = (__bf16*)(ws + (1 << 15));                   // [P][K][H] bf16, 4MB
  __bf16* w2t      = (__bf16*)(ws + (1 << 15) + (4 << 20));       // [P][H][K] bf16, 4MB
  __bf16* hidb     = (__bf16*)(ws + (1 << 15) + (8 << 20));       // [T][H]  bf16, 16MB

  k_router<<<B_, 256, 0, stream>>>(hidden, head_w, head_b, probs_ws);
  k_b1eff<<<P_, 256, 0, stream>>>(embeds, w1, b1, b1e_ws);
  k_transpose_bf16<<<(P_ * H_ * K_) / 256, 256, 0, stream>>>(w1, w1t, H_, K_);
  k_transpose_bf16<<<(P_ * K_ * H_) / 256, 256, 0, stream>>>(w2, w2t, K_, H_);
  k_hid_bf16<<<(T_ * H_) / (256 * 4), 256, 0, stream>>>(hidden, hidb);
  k_mlp<<<T_ / 32, 256, 0, stream>>>(hidb, w1t, w2t, b1e_ws, b2, probs_ws, out);
}
